// PixelQueryNet_65128884077193
// MI455X (gfx1250) — matrix operand order repaired and checked by reading, not compiled, not run on hardware
//
#include <hip/hip_runtime.h>
#include <hip/hip_bf16.h>

typedef __attribute__((ext_vector_type(16))) _Float16 v16h;
typedef __attribute__((ext_vector_type(8)))  _Float16 v8h;
typedef __attribute__((ext_vector_type(8)))  float    v8f;
typedef int v4i __attribute__((vector_size(4 * sizeof(int))));

#define NPARAM 12995
#define PLANE  4096           // 64*64 low-res cells per param plane
#define NC     8              // cells (waves) per workgroup

__device__ __forceinline__ v16h cat8(v8h lo, v8h hi) {
  return __builtin_shufflevector(lo, hi, 0,1,2,3,4,5,6,7,8,9,10,11,12,13,14,15);
}

// ---- gfx1250 async global->LDS (ASYNCcnt-tracked), with safe fallback -------
__device__ __forceinline__ void async_g2l_b128(void* lds, const void* g) {
#if __has_builtin(__builtin_amdgcn_global_load_async_to_lds_b128)
  __builtin_amdgcn_global_load_async_to_lds_b128(
      (__attribute__((address_space(1))) v4i*)(uintptr_t)g,
      (__attribute__((address_space(3))) v4i*)(uintptr_t)lds, 0, 0);
#else
  *(float4*)lds = *(const float4*)g;
#endif
}
__device__ __forceinline__ void async_wait_all() {
#if __has_builtin(__builtin_amdgcn_s_wait_asynccnt)
  __builtin_amdgcn_s_wait_asynccnt(0);
#elif __has_builtin(__builtin_amdgcn_global_load_async_to_lds_b128)
  asm volatile("s_wait_asynccnt 0x0" ::: "memory");
#endif
}

// bias rows (f32) -> LDS [o][cell] via async DMA; rows2 = 2*rows
__device__ __forceinline__ void stage_bias_async(int tid, float* dst, const float* gB,
                                                 int rows2) {
  for (int idx = tid; idx < rows2; idx += 256) {
    int h = idx & 1, o = idx >> 1;
    async_g2l_b128(dst + o * NC + h * 4, gB + (size_t)o * PLANE + h * 4);
  }
}

// One middle layer (64x64 @ 64x64, +bias, leaky) for this wave's cell.
// SK=1: additionally software-pipeline the *next* layer's weight staging
// (8 float4 global loads issued before each nt-tile's WMMAs, converted and
// stored to LDS after — LOADcnt drains overlap the DScnt-bound compute).
template <int SK>
__device__ __forceinline__ void mid_layer(int tid, int wave, int m, int hi,
                                          const _Float16* actIn, const _Float16* wcur,
                                          const float* bcur, _Float16* actOut,
                                          const float* gW, _Float16* wnext) {
  const _Float16* in = actIn + wave * 4096;
  const _Float16* w  = wcur  + wave * 4096;
  _Float16*       og = actOut + wave * 4096;

  #pragma unroll
  for (int nt = 0; nt < 4; ++nt) {
    float4 pend[8];
    if (SK) {
      #pragma unroll
      for (int q = 0; q < 8; ++q) {
        int idx = tid + (nt * 8 + q) * 256;       // 32 float4 per thread total
        int h = idx & 1, p = idx >> 1;            // p = i*64 + o
        pend[q] = *(const float4*)(gW + (size_t)p * PLANE + h * 4);
      }
    }

    const float bv = bcur[(nt * 16 + m) * NC + wave];
    v8f acc[4];
    #pragma unroll
    for (int mt = 0; mt < 4; ++mt)
      #pragma unroll
      for (int r = 0; r < 8; ++r) acc[mt][r] = bv;

    #pragma unroll
    for (int kt = 0; kt < 2; ++kt) {
      const _Float16* wp = w + (nt * 16 + m) * 64 + kt * 32 + hi * 8;
      v16h bf = cat8(*(const v8h*)wp, *(const v8h*)(wp + 16));
      #pragma unroll
      for (int mt = 0; mt < 4; ++mt) {
        const _Float16* ip = in + (mt * 16 + m) * 64 + kt * 32 + hi * 8;
        v16h af = cat8(*(const v8h*)ip, *(const v8h*)(ip + 16));
        acc[mt] = __builtin_amdgcn_wmma_f32_16x16x32_f16(false, af, false, bf,
                                                         (short)0, acc[mt], false, false);
      }
    }
    #pragma unroll
    for (int mt = 0; mt < 4; ++mt)
      #pragma unroll
      for (int r = 0; r < 8; ++r) {
        float v = acc[mt][r];
        v = fmaxf(v, 0.01f * v);                  // leaky_relu(0.01) == max(x, 0.01x)
        og[(mt * 16 + hi * 8 + r) * 64 + nt * 16 + m] = (_Float16)v;
      }

    if (SK) {
      #pragma unroll
      for (int q = 0; q < 8; ++q) {
        int idx = tid + (nt * 8 + q) * 256;
        int h = idx & 1, p = idx >> 1;
        int i = p >> 6, o = p & 63;
        _Float16* dst = wnext + (h * 4) * 4096 + o * 64 + i;
        dst[0]        = (_Float16)pend[q].x;
        dst[4096]     = (_Float16)pend[q].y;
        dst[2 * 4096] = (_Float16)pend[q].z;
        dst[3 * 4096] = (_Float16)pend[q].w;
      }
    }
  }
}

__global__ __launch_bounds__(256)
void pixelquery_mlp_kernel(const float* __restrict__ gp,   // lr_params (2,12995,64,64)
                           float* __restrict__ out)        // (2,3,512,512)
{
  constexpr int BOFF[5] = {0, 320, 4480, 8640, 12800};
  constexpr int WOFF[5] = {64, 384, 4544, 8704, 12803};

  extern __shared__ __align__(16) char smem_raw[];
  _Float16* actA = (_Float16*)smem_raw;           // NC*4096 f16, [px][ch]
  _Float16* actB = actA + NC * 4096;              // NC*4096 f16 (w0s overlays head)
  _Float16* wTa  = actB + NC * 4096;              // NC*4096 f16, [cell][o*64+i]
  _Float16* wTb  = wTa  + NC * 4096;              // NC*4096 f16
  float*    bSa  = (float*)(wTb + NC * 4096);     // 64*NC f32, [o][cell]  (L1, L3)
  float*    bSb  = bSa + 64 * NC;                 //                      (L2, L4)
  float*    bS0  = bSb + 64 * NC;                 //                      (L0)

  const int tid  = threadIdx.x;
  const int lane = tid & 31;
  const int wave = tid >> 5;
  const int m    = lane & 15;
  const int hi   = lane >> 4;

  const int wg  = blockIdx.x;                     // 1024 = 2 * 64 * 8
  const int b   = wg >> 9;
  const int rem = wg & 511;
  const int y   = rem >> 3;
  const int x0  = (rem & 7) * NC;

  const size_t batch   = (size_t)b * NPARAM * PLANE;
  const int    cellrow = y * 64 + x0;
  const float* gpc     = gp + batch + cellrow;    // cell-row base

  // ========== P0: stage L0 (async f32) + L1 (f16 transposed) ==========
  float* w0s = (float*)actB;                      // [p][cx], 256*NC f32 (8KB)
  stage_bias_async(tid, bS0, gpc + (size_t)BOFF[0] * PLANE, 128);
  stage_bias_async(tid, bSa, gpc + (size_t)BOFF[1] * PLANE, 128);
  for (int idx = tid; idx < 256 * 2; idx += 256) {
    int h = idx & 1, p = idx >> 1;
    async_g2l_b128(w0s + p * NC + h * 4, gpc + (size_t)(WOFF[0] + p) * PLANE + h * 4);
  }
  {
    const float* gW1 = gpc + (size_t)WOFF[1] * PLANE;
    for (int idx = tid; idx < 2 * 4096; idx += 256) {
      int h = idx & 1, p = idx >> 1;
      int i = p >> 6, o = p & 63;
      const float4 v = *(const float4*)(gW1 + (size_t)p * PLANE + h * 4);
      _Float16* dst = wTa + (h * 4) * 4096 + o * 64 + i;
      dst[0]        = (_Float16)v.x;
      dst[4096]     = (_Float16)v.y;
      dst[2 * 4096] = (_Float16)v.z;
      dst[3 * 4096] = (_Float16)v.w;
    }
  }
  async_wait_all();
  __syncthreads();

  // ========== P1: compute L0 -> actA, fused with staging L2 -> wTb ==========
  {
    stage_bias_async(tid, bSb, gpc + (size_t)BOFF[2] * PLANE, 128);
    const float* gW2 = gpc + (size_t)WOFF[2] * PLANE;

    float wa[4], wb[4];
    #pragma unroll
    for (int i = 0; i < 4; ++i) {
      wa[i] = w0s[(i * 64 + lane) * NC + wave];
      wb[i] = w0s[(i * 64 + lane + 32) * NC + wave];
    }
    const float ba = bS0[lane * NC + wave], bb = bS0[(lane + 32) * NC + wave];
    _Float16* ao = actA + wave * 4096;

    #pragma unroll
    for (int seg = 0; seg < 4; ++seg) {
      float4 pend[8];
      #pragma unroll
      for (int q = 0; q < 8; ++q) {
        int idx = tid + (seg * 8 + q) * 256;
        int h = idx & 1, p = idx >> 1;
        pend[q] = *(const float4*)(gW2 + (size_t)p * PLANE + h * 4);
      }
      #pragma unroll
      for (int ps = 0; ps < 16; ++ps) {
        int px = seg * 16 + ps;
        float xs, xc, ys, yc;
        __sincosf((float)(px & 7)  * 0.7853981633974483f, &xs, &xc);
        __sincosf((float)(px >> 3) * 0.7853981633974483f, &ys, &yc);
        float v0 = ba + xc * wa[0] + xs * wa[1] + yc * wa[2] + ys * wa[3];
        float v1 = bb + xc * wb[0] + xs * wb[1] + yc * wb[2] + ys * wb[3];
        ao[px * 64 + lane]      = (_Float16)fmaxf(v0, 0.01f * v0);
        ao[px * 64 + lane + 32] = (_Float16)fmaxf(v1, 0.01f * v1);
      }
      #pragma unroll
      for (int q = 0; q < 8; ++q) {
        int idx = tid + (seg * 8 + q) * 256;
        int h = idx & 1, p = idx >> 1;
        int i = p >> 6, o = p & 63;
        _Float16* dst = wTb + (h * 4) * 4096 + o * 64 + i;
        dst[0]        = (_Float16)pend[q].x;
        dst[4096]     = (_Float16)pend[q].y;
        dst[2 * 4096] = (_Float16)pend[q].z;
        dst[3 * 4096] = (_Float16)pend[q].w;
      }
    }
    async_wait_all();
    __syncthreads();
  }

  // ========== P2: compute L1 (actA x wTa -> actB) ==========
  mid_layer<0>(tid, wave, m, hi, actA, wTa, bSa, actB, nullptr, nullptr);
  __syncthreads();

  // ========== P3: compute L2 (actB x wTb -> actA) + stage L3 -> wTa ==========
  stage_bias_async(tid, bSa, gpc + (size_t)BOFF[3] * PLANE, 128);
  mid_layer<1>(tid, wave, m, hi, actB, wTb, bSb, actA,
               gpc + (size_t)WOFF[3] * PLANE, wTa);
  async_wait_all();
  __syncthreads();

  // ========== P4: compute L3 (actA x wTa -> actB) + stage L4 -> wTb ==========
  {
    // L4 bias rows o<3 via async (6 b128 ops across the WG)
    for (int idx = tid; idx < 6; idx += 256) {
      int h = idx & 1, o = idx >> 1;
      async_g2l_b128(bSb + o * NC + h * 4,
                     gpc + (size_t)(BOFF[4] + o) * PLANE + h * 4);
    }
    // issue L4 weight loads (<=2 float4 per thread) before the L3 compute
    const float* gW4 = gpc + (size_t)WOFF[4] * PLANE;
    float4 wl[2];
    #pragma unroll
    for (int q = 0; q < 2; ++q) {
      int idx = tid + q * 256;
      if (idx < 2 * 192) {
        int h = idx & 1, p = idx >> 1;
        wl[q] = *(const float4*)(gW4 + (size_t)p * PLANE + h * 4);
      }
    }

    mid_layer<0>(tid, wave, m, hi, actA, wTa, bSa, actB, nullptr, nullptr);

    // drain L4 weights -> wTb [cell][o*64+i], o<3
    #pragma unroll
    for (int q = 0; q < 2; ++q) {
      int idx = tid + q * 256;
      if (idx < 2 * 192) {
        int h = idx & 1, p = idx >> 1;                 // p = i*3 + o
        int i = p / 3, o = p % 3;
        _Float16* dst = wTb + (h * 4) * 4096 + o * 64 + i;
        dst[0]        = (_Float16)wl[q].x;
        dst[4096]     = (_Float16)wl[q].y;
        dst[2 * 4096] = (_Float16)wl[q].z;
        dst[3 * 4096] = (_Float16)wl[q].w;
      }
    }
    for (int idx = tid; idx < NC * 832; idx += 256) {  // zero rows o=3..15
      int cx = idx & 7, r = idx >> 3;
      wTb[cx * 4096 + 192 + r] = (_Float16)0.f;
    }
    for (int idx = tid; idx < 104; idx += 256)         // zero bias rows 3..15
      bSb[24 + idx] = 0.f;
    async_wait_all();
    __syncthreads();
  }

  // ========== P5: compute L4 (actB x wTb, tanh) -> global ==========
  {
    const _Float16* in = actB + wave * 4096;
    const _Float16* w  = wTb  + wave * 4096;
    const float bv = bSb[m * NC + wave];
    v8f acc[4];
    #pragma unroll
    for (int mt = 0; mt < 4; ++mt)
      #pragma unroll
      for (int r = 0; r < 8; ++r) acc[mt][r] = bv;

    #pragma unroll
    for (int kt = 0; kt < 2; ++kt) {
      const _Float16* wp = w + m * 64 + kt * 32 + hi * 8;
      v16h bf = cat8(*(const v8h*)wp, *(const v8h*)(wp + 16));
      #pragma unroll
      for (int mt = 0; mt < 4; ++mt) {
        const _Float16* ip = in + (mt * 16 + m) * 64 + kt * 32 + hi * 8;
        v16h af = cat8(*(const v8h*)ip, *(const v8h*)(ip + 16));
        acc[mt] = __builtin_amdgcn_wmma_f32_16x16x32_f16(false, af, false, bf,
                                                         (short)0, acc[mt], false, false);
      }
    }

    if (m < 3) {
      const int xcol0 = (x0 + wave) * 8;
      #pragma unroll
      for (int mt = 0; mt < 4; ++mt)
        #pragma unroll
        for (int r = 0; r < 8; ++r) {
          int px = mt * 16 + hi * 8 + r;
          float v = tanhf(acc[mt][r]);
          out[(((size_t)b * 3 + m) * 512 + (y * 8 + (px >> 3))) * 512 + xcol0 + (px & 7)] = v;
        }
    }
  }
}

extern "C" void kernel_launch(void* const* d_in, const int* in_sizes, int n_in,
                              void* d_out, int out_size, void* d_ws, size_t ws_size,
                              hipStream_t stream) {
  (void)in_sizes; (void)n_in; (void)d_ws; (void)ws_size; (void)out_size;
  // d_in[0] = highres (unused: reference overwrites it with the positional
  // encoding, which we synthesize on the fly), d_in[1] = lr_params.
  const float* lrp = (const float*)d_in[1];
  float* out = (float*)d_out;

  const size_t shmem = (size_t)4 * NC * 4096 * sizeof(_Float16)   // actA/actB/wTa/wTb
                     + (size_t)3 * 64 * NC * sizeof(float);       // bSa/bSb/bS0
  // 2 batches * 64 rows * 8 x-chunks = 1024 workgroups of 8 waves (8 cells each)
  pixelquery_mlp_kernel<<<dim3(1024), dim3(256), shmem, stream>>>(lrp, out);
}
